// AttnBlock_49349174231479
// MI455X (gfx1250) — matrix operand order repaired and checked by reading, not compiled, not run on hardware
//
#include <hip/hip_runtime.h>
#include <hip/hip_bf16.h>
#include <math.h>

// ---------------------------------------------------------------- constants
#define BATCH  4
#define CDIM   512
#define HWSZ   4096
#define NGROUP 32
#define CPG    (CDIM / NGROUP)   // 16 channels per group

typedef __attribute__((ext_vector_type(16))) __bf16 bf16x16;
typedef __attribute__((ext_vector_type(8)))  __bf16 bf16x8;
typedef __attribute__((ext_vector_type(8)))  float  f32x8;
typedef __attribute__((ext_vector_type(4)))  unsigned int u32x4;
typedef __attribute__((ext_vector_type(8)))  int i32x8;
typedef __attribute__((ext_vector_type(4)))  int i32x4;

union FragAB { bf16x16 v; bf16x8 h[2]; unsigned short u[16]; };
union FragC  { f32x8 v; float f[8]; };

__device__ __forceinline__ unsigned short f2bf(float x) {
  union { float f; unsigned u; } c; c.f = x;
  unsigned u = c.u;
  u += 0x7FFFu + ((u >> 16) & 1u);   // round-to-nearest-even
  return (unsigned short)(u >> 16);
}

// A fragment: 16x32 bf16: lanes 0-15 hold K[k..k+15] of row m,
// lanes 16-31 hold K[k+16..k+31].
__device__ __forceinline__ void loadA(const unsigned short* base, int lda,
                                      int m0, int k0, int lane, FragAB& f) {
  int row = m0 + (lane & 15);
  int kb  = k0 + ((lane >> 4) << 4);
  const unsigned short* p = base + (size_t)row * lda + kb;
  f.h[0] = *(const bf16x8*)(p);
  f.h[1] = *(const bf16x8*)(p + 8);
}

// B fragment: 32x16 bf16 from BT[N, K] (K contiguous). Lane = column n0+(lane&15);
// lanes 0-15: K[k0..k0+15], lanes 16-31: K[k0+16..k0+31].
__device__ __forceinline__ void loadB(const unsigned short* base, int ldb,
                                      int n0, int k0, int lane, FragAB& f) {
  int row = n0 + (lane & 15);
  int kb  = k0 + ((lane >> 4) << 4);
  const unsigned short* p = base + (size_t)row * ldb + kb;
  f.h[0] = *(const bf16x8*)(p);
  f.h[1] = *(const bf16x8*)(p + 8);
}

// Same as loadB but source rows are f32 (softmax probabilities), convert on the fly.
__device__ __forceinline__ void loadB_f32(const float* base, int ldb,
                                          int n0, int k0, int lane, FragAB& f) {
  int row = n0 + (lane & 15);
  int kb  = k0 + ((lane >> 4) << 4);
  const float* p = base + (size_t)row * ldb + kb;
#pragma unroll
  for (int i = 0; i < 16; i += 4) {
    float4 t = *(const float4*)(p + i);
    f.u[i + 0] = f2bf(t.x); f.u[i + 1] = f2bf(t.y);
    f.u[i + 2] = f2bf(t.z); f.u[i + 3] = f2bf(t.w);
  }
}

// ---------------------------------------------------------------- TDM: 2D bf16 tile -> LDS
// Builds a Tensor DMA Descriptor (ISA cdna5 §8.3/8.4) for a tileRows x tileK tile of
// 2-byte elements with row stride rowStrideElems, and issues TENSOR_LOAD_TO_LDS.
// Tracked by TENSORcnt. All arguments are workgroup-uniform.
__device__ __forceinline__ void tdm_load_2d(unsigned ldsOff, const void* gaddr,
                                            unsigned tileK, unsigned tileRows,
                                            unsigned rowStrideElems) {
  unsigned long long ga = (unsigned long long)gaddr;
  u32x4 g0;
  g0[0] = 1u;                                               // count=1, user-mode D#
  g0[1] = ldsOff;                                           // lds_addr (bytes)
  g0[2] = (unsigned)(ga & 0xFFFFFFFFu);                     // global_addr[31:0]
  g0[3] = (unsigned)((ga >> 32) & 0x01FFFFFFu) | (2u << 30);// global_addr[56:32], type=2
  i32x8 g1;
  g1[0] = (int)(1u << 16);                                  // data_size=1 (2 bytes)
  g1[1] = (int)((tileK & 0xFFFFu) << 16);                   // tensor_dim0[15:0] @ 63:48
  g1[2] = (int)(((tileK >> 16) & 0xFFFFu) |                 // tensor_dim0[31:16]
                ((tileRows & 0xFFFFu) << 16));              // tensor_dim1[15:0]
  g1[3] = (int)(((tileRows >> 16) & 0xFFFFu) |              // tensor_dim1[31:16]
                ((tileK & 0xFFFFu) << 16));                 // tile_dim0
  g1[4] = (int)(tileRows & 0xFFFFu);                        // tile_dim1 (tile_dim2=0)
  g1[5] = (int)rowStrideElems;                              // tensor_dim0_stride[31:0]
  g1[6] = 0;
  g1[7] = 0;
  i32x4 gz = {0, 0, 0, 0};
#if defined(__clang_major__) && (__clang_major__ >= 23)
  i32x8 gz8 = {0, 0, 0, 0, 0, 0, 0, 0};
  __builtin_amdgcn_tensor_load_to_lds(g0, g1, gz, gz, gz8, 0);
#else
  __builtin_amdgcn_tensor_load_to_lds(g0, g1, gz, gz, 0);
#endif
}

// ---------------------------------------------------------------- fp32 -> bf16 weights
__global__ void k_cvt_bf16(const float* __restrict__ in,
                           unsigned short* __restrict__ out, int n) {
  int i = blockIdx.x * blockDim.x + threadIdx.x;
  if (i < n) out[i] = f2bf(in[i]);
}

// ---------------------------------------------------------------- GroupNorm -> hT bf16 [B, HW, C]
__global__ void k_groupnorm(const float* __restrict__ x,
                            const float* __restrict__ gamma,
                            const float* __restrict__ beta,
                            unsigned short* __restrict__ hT) {
  int b = blockIdx.x >> 5;
  int g = blockIdx.x & 31;
  const int n = CPG * HWSZ;                       // 65536 contiguous floats per group
  const float* px = x + ((size_t)b * CDIM + g * CPG) * HWSZ;
  int t = threadIdx.x;

  float s = 0.f, ss = 0.f;
  for (int i = t * 4; i < n; i += 256 * 4) {
    float4 v = *(const float4*)(px + i);
    s  += v.x + v.y + v.z + v.w;
    ss += v.x * v.x + v.y * v.y + v.z * v.z + v.w * v.w;
  }
  __shared__ float rs[256], rss[256];
  rs[t] = s; rss[t] = ss; __syncthreads();
  for (int o = 128; o > 0; o >>= 1) {
    if (t < o) { rs[t] += rs[t + o]; rss[t] += rss[t + o]; }
    __syncthreads();
  }
  float mu   = rs[0] * (1.f / 65536.f);
  float var  = rss[0] * (1.f / 65536.f) - mu * mu;
  float rstd = rsqrtf(var + 1e-6f);

  unsigned short* ph = hT + (size_t)b * HWSZ * CDIM;
  for (int i = t; i < n; i += 256) {
    int cl = i >> 12;            // local channel
    int p  = i & (HWSZ - 1);     // spatial index
    int c  = g * CPG + cl;
    float v = (px[i] - mu) * rstd * gamma[c] + beta[c];
    ph[(size_t)p * CDIM + c] = f2bf(v);
  }
}

// ---------------------------------------------------------------- projection GEMM (q/k/v)
// C[m,n] = sum_k W[m,k] * hT[n,k];  M=512, N=4096, K=512; wave tile 64x16.
__global__ void k_gemm_qkv(const unsigned short* __restrict__ Wbf,
                           const unsigned short* __restrict__ hT,
                           const float* __restrict__ bias,
                           unsigned short* __restrict__ out,
                           float scale, int storeT) {
  const int M = CDIM, N = HWSZ, K = CDIM;
  int wave = blockIdx.x * (blockDim.x >> 5) + (threadIdx.x >> 5);
  int lane = threadIdx.x & 31;
  const int nT = N / 16, mG = M / 64, perB = nT * mG;
  int b  = wave / perB;
  int r  = wave % perB;
  int n0 = (r % nT) * 16;
  int m0 = (r / nT) * 64;
  if (b >= BATCH) return;

  const unsigned short* BT = hT + (size_t)b * N * K;
  FragC acc[4];
  for (int i = 0; i < 4; i++)
    for (int j = 0; j < 8; j++) acc[i].f[j] = 0.f;

  for (int k0 = 0; k0 < K; k0 += 32) {
    FragAB fb; loadB(BT, K, n0, k0, lane, fb);
#pragma unroll
    for (int ms = 0; ms < 4; ms++) {
      FragAB fa; loadA(Wbf, K, m0 + ms * 16, k0, lane, fa);
      acc[ms].v = __builtin_amdgcn_wmma_f32_16x16x32_bf16(
          false, fa.v, false, fb.v, (short)0, acc[ms].v, false, false);
    }
  }

  int ncol  = n0 + (lane & 15);
  int rbase = (lane >> 4) << 3;
  if (storeT) {
    unsigned short* o = out + (size_t)b * N * M;
#pragma unroll
    for (int ms = 0; ms < 4; ms++)
#pragma unroll
      for (int rr = 0; rr < 8; rr++) {
        int m = m0 + ms * 16 + rbase + rr;
        o[(size_t)ncol * M + m] = f2bf((acc[ms].f[rr] + bias[m]) * scale);
      }
  } else {
    unsigned short* o = out + (size_t)b * M * N;
#pragma unroll
    for (int ms = 0; ms < 4; ms++)
#pragma unroll
      for (int rr = 0; rr < 8; rr++) {
        int m = m0 + ms * 16 + rbase + rr;
        o[(size_t)m * N + ncol] = f2bf(acc[ms].f[rr] + bias[m]);
      }
  }
}

// ---------------------------------------------------------------- S = qT * kT^T  (scores, f32)
// M=N=4096, K=512. Block = 8 waves computing a 128x64 tile; A/B tiles staged into
// double-buffered LDS by the Tensor Data Mover, pipelined against WMMA consumption.
#define AT_TM 128          // block tile M
#define AT_TN 64           // block tile N
#define AT_TK 32           // k step
__global__ void k_gemm_attn(const unsigned short* __restrict__ qT,
                            const unsigned short* __restrict__ kT,
                            float* __restrict__ S) {
  const int M = HWSZ, N = HWSZ, K = CDIM;
  __shared__ __align__(16) unsigned short smA[2][AT_TM * AT_TK]; // 2 x 8 KB
  __shared__ __align__(16) unsigned short smB[2][AT_TN * AT_TK]; // 2 x 4 KB

  int wid  = threadIdx.x >> 5;
  int lane = threadIdx.x & 31;
  const int mBlkCnt = M / AT_TM;             // 32
  const int nBlkCnt = N / AT_TN;             // 64
  const int perB = mBlkCnt * nBlkCnt;        // 2048
  int b    = blockIdx.x / perB;
  int r    = blockIdx.x % perB;
  int mBlk = (r / nBlkCnt) * AT_TM;
  int nBlk = (r % nBlkCnt) * AT_TN;
  if (b >= BATCH) return;

  const unsigned short* gA = qT + (size_t)b * M * K + (size_t)mBlk * K; // [128 rows, K]
  const unsigned short* gB = kT + (size_t)b * N * K + (size_t)nBlk * K; // [64 rows, K]

  int wm = (wid & 1) * 64;        // wave M offset within block tile
  int wn = (wid >> 1) * 16;       // wave N offset within block tile

  FragC acc[4];
  for (int i = 0; i < 4; i++)
    for (int j = 0; j < 8; j++) acc[i].f[j] = 0.f;

  const int NK = K / AT_TK;       // 16 k-steps
  // prologue: wave 0 DMAs tile 0 into buffer 0
  if (wid == 0) {
    tdm_load_2d((unsigned)(size_t)&smA[0][0], gA, AT_TK, AT_TM, K);
    tdm_load_2d((unsigned)(size_t)&smB[0][0], gB, AT_TK, AT_TN, K);
  }

  for (int ks = 0; ks < NK; ks++) {
    int cur = ks & 1;
    if (wid == 0) {
      if (ks + 1 < NK) {
        int nxt = cur ^ 1;
        // stream next tile while current one is consumed
        tdm_load_2d((unsigned)(size_t)&smA[nxt][0], gA + (ks + 1) * AT_TK, AT_TK, AT_TM, K);
        tdm_load_2d((unsigned)(size_t)&smB[nxt][0], gB + (ks + 1) * AT_TK, AT_TK, AT_TN, K);
        __builtin_amdgcn_s_wait_tensorcnt(2);   // current tile's 2 DMAs retired
      } else {
        __builtin_amdgcn_s_wait_tensorcnt(0);
      }
    }
    __syncthreads();

    // fragments from LDS (tile rows packed contiguously, ld = AT_TK)
    FragAB fb;
    {
      int row = wn + (lane & 15);
      int kb  = (lane >> 4) << 4;
      const unsigned short* p = &smB[cur][row * AT_TK + kb];
      fb.h[0] = *(const bf16x8*)(p);
      fb.h[1] = *(const bf16x8*)(p + 8);
    }
#pragma unroll
    for (int ms = 0; ms < 4; ms++) {
      FragAB fa;
      int row = wm + ms * 16 + (lane & 15);
      int kb  = (lane >> 4) << 4;
      const unsigned short* p = &smA[cur][row * AT_TK + kb];
      fa.h[0] = *(const bf16x8*)(p);
      fa.h[1] = *(const bf16x8*)(p + 8);
      acc[ms].v = __builtin_amdgcn_wmma_f32_16x16x32_bf16(
          false, fa.v, false, fb.v, (short)0, acc[ms].v, false, false);
    }
    __syncthreads();
  }

  float* o  = S + (size_t)b * M * N;
  int ncol  = nBlk + wn + (lane & 15);
  int rbase = (lane >> 4) << 3;
#pragma unroll
  for (int ms = 0; ms < 4; ms++)
#pragma unroll
    for (int rr = 0; rr < 8; rr++) {
      int m = mBlk + wm + ms * 16 + rbase + rr;
      o[(size_t)m * N + ncol] = acc[ms].f[rr];
    }
}

// ---------------------------------------------------------------- row softmax, in place
__global__ void k_softmax(float* __restrict__ S) {
  const int N = HWSZ;
  float* p = S + (size_t)blockIdx.x * N;
  int t = threadIdx.x;
  __shared__ float red[256];

  float m = -3.4e38f;
  for (int i = t; i < N; i += 256) m = fmaxf(m, p[i]);
  red[t] = m; __syncthreads();
  for (int o = 128; o > 0; o >>= 1) {
    if (t < o) red[t] = fmaxf(red[t], red[t + o]);
    __syncthreads();
  }
  m = red[0]; __syncthreads();

  float s = 0.f;
  for (int i = t; i < N; i += 256) {
    float e = __expf(p[i] - m);
    p[i] = e; s += e;
  }
  red[t] = s; __syncthreads();
  for (int o = 128; o > 0; o >>= 1) {
    if (t < o) red[t] += red[t + o];
    __syncthreads();
  }
  float inv = 1.f / red[0];
  for (int i = t; i < N; i += 256) p[i] *= inv;
}

// ---------------------------------------------------------------- O = v * P^T -> oT bf16 [HW, C]
// M=512, N=4096, K=4096;  A = v[c, j] bf16, BT = P[i, j] f32 (converted per fragment)
__global__ void k_gemm_o(const unsigned short* __restrict__ v,
                         const float* __restrict__ P,
                         unsigned short* __restrict__ oT) {
  const int M = CDIM, N = HWSZ, K = HWSZ;
  int wave = blockIdx.x * (blockDim.x >> 5) + (threadIdx.x >> 5);
  int lane = threadIdx.x & 31;
  const int nT = N / 16, mG = M / 64, perB = nT * mG;
  int b  = wave / perB;
  int r  = wave % perB;
  int n0 = (r % nT) * 16;
  int m0 = (r / nT) * 64;
  if (b >= BATCH) return;

  const unsigned short* A  = v + (size_t)b * M * K;
  const float*          BT = P + (size_t)b * N * K;
  FragC acc[4];
  for (int i = 0; i < 4; i++)
    for (int j = 0; j < 8; j++) acc[i].f[j] = 0.f;

  for (int k0 = 0; k0 < K; k0 += 32) {
    FragAB fb; loadB_f32(BT, K, n0, k0, lane, fb);
#pragma unroll
    for (int ms = 0; ms < 4; ms++) {
      FragAB fa; loadA(A, K, m0 + ms * 16, k0, lane, fa);
      acc[ms].v = __builtin_amdgcn_wmma_f32_16x16x32_bf16(
          false, fa.v, false, fb.v, (short)0, acc[ms].v, false, false);
    }
  }

  unsigned short* o = oT + (size_t)b * N * M;
  int ncol  = n0 + (lane & 15);
  int rbase = (lane >> 4) << 3;
#pragma unroll
  for (int ms = 0; ms < 4; ms++)
#pragma unroll
    for (int rr = 0; rr < 8; rr++) {
      int m = m0 + ms * 16 + rbase + rr;
      o[(size_t)ncol * M + m] = f2bf(acc[ms].f[rr]);
    }
}

// ---------------------------------------------------------------- y = x + Wp * o + bp
// M=512, N=4096, K=512;  BT = oT[n, k]
__global__ void k_gemm_proj(const unsigned short* __restrict__ Wp,
                            const unsigned short* __restrict__ oT,
                            const float* __restrict__ bias,
                            const float* __restrict__ x,
                            float* __restrict__ y) {
  const int M = CDIM, N = HWSZ, K = CDIM;
  int wave = blockIdx.x * (blockDim.x >> 5) + (threadIdx.x >> 5);
  int lane = threadIdx.x & 31;
  const int nT = N / 16, mG = M / 64, perB = nT * mG;
  int b  = wave / perB;
  int r  = wave % perB;
  int n0 = (r % nT) * 16;
  int m0 = (r / nT) * 64;
  if (b >= BATCH) return;

  const unsigned short* BT = oT + (size_t)b * N * K;
  FragC acc[4];
  for (int i = 0; i < 4; i++)
    for (int j = 0; j < 8; j++) acc[i].f[j] = 0.f;

  for (int k0 = 0; k0 < K; k0 += 32) {
    FragAB fb; loadB(BT, K, n0, k0, lane, fb);
#pragma unroll
    for (int ms = 0; ms < 4; ms++) {
      FragAB fa; loadA(Wp, K, m0 + ms * 16, k0, lane, fa);
      acc[ms].v = __builtin_amdgcn_wmma_f32_16x16x32_bf16(
          false, fa.v, false, fb.v, (short)0, acc[ms].v, false, false);
    }
  }

  int ncol  = n0 + (lane & 15);
  int rbase = (lane >> 4) << 3;
#pragma unroll
  for (int ms = 0; ms < 4; ms++)
#pragma unroll
    for (int rr = 0; rr < 8; rr++) {
      int m = m0 + ms * 16 + rbase + rr;
      size_t idx = (size_t)b * M * N + (size_t)m * N + ncol;
      y[idx] = x[idx] + acc[ms].f[rr] + bias[m];
    }
}

// ---------------------------------------------------------------- launcher
extern "C" void kernel_launch(void* const* d_in, const int* in_sizes, int n_in,
                              void* d_out, int out_size, void* d_ws, size_t ws_size,
                              hipStream_t stream) {
  const float* x     = (const float*)d_in[0];
  const float* gamma = (const float*)d_in[1];
  const float* beta  = (const float*)d_in[2];
  const float* Wq    = (const float*)d_in[3];
  const float* bq    = (const float*)d_in[4];
  const float* Wk    = (const float*)d_in[5];
  const float* bk    = (const float*)d_in[6];
  const float* Wv    = (const float*)d_in[7];
  const float* bv    = (const float*)d_in[8];
  const float* Wp    = (const float*)d_in[9];
  const float* bp    = (const float*)d_in[10];
  float* out = (float*)d_out;

  // workspace carve-up (256B aligned)
  char* w = (char*)d_ws;
  auto carve = [&](size_t bytes) {
    void* p = (void*)w;
    w += (bytes + 255) & ~(size_t)255;
    return p;
  };
  const size_t act_e = (size_t)BATCH * HWSZ * CDIM;           // 8.4M elements
  unsigned short* hT  = (unsigned short*)carve(act_e * 2);    // [B, HW, C] bf16
  unsigned short* qT  = (unsigned short*)carve(act_e * 2);
  unsigned short* kT  = (unsigned short*)carve(act_e * 2);
  unsigned short* vR  = (unsigned short*)carve(act_e * 2);    // [B, C, HW] bf16
  unsigned short* oT  = (unsigned short*)carve(act_e * 2);
  unsigned short* Wqb = (unsigned short*)carve((size_t)CDIM * CDIM * 2);
  unsigned short* Wkb = (unsigned short*)carve((size_t)CDIM * CDIM * 2);
  unsigned short* Wvb = (unsigned short*)carve((size_t)CDIM * CDIM * 2);
  unsigned short* Wpb = (unsigned short*)carve((size_t)CDIM * CDIM * 2);
  float* S = (float*)carve((size_t)BATCH * HWSZ * HWSZ * 4);  // 256 MB scores

  const int nW = CDIM * CDIM;
  k_cvt_bf16<<<(nW + 255) / 256, 256, 0, stream>>>(Wq, Wqb, nW);
  k_cvt_bf16<<<(nW + 255) / 256, 256, 0, stream>>>(Wk, Wkb, nW);
  k_cvt_bf16<<<(nW + 255) / 256, 256, 0, stream>>>(Wv, Wvb, nW);
  k_cvt_bf16<<<(nW + 255) / 256, 256, 0, stream>>>(Wp, Wpb, nW);

  k_groupnorm<<<BATCH * NGROUP, 256, 0, stream>>>(x, gamma, beta, hT);

  const float scale = 1.0f / sqrtf((float)CDIM);
  // waves = B * (512/64) * (4096/16) = 8192 -> 1024 blocks of 8 waves
  k_gemm_qkv<<<1024, 256, 0, stream>>>(Wqb, hT, bq, qT, scale, 1);
  k_gemm_qkv<<<1024, 256, 0, stream>>>(Wkb, hT, bk, kT, 1.0f, 1);
  k_gemm_qkv<<<1024, 256, 0, stream>>>(Wvb, hT, bv, vR, 1.0f, 0);

  // blocks = B * (4096/128) * (4096/64) = 8192, each 8 waves on a 128x64 tile
  k_gemm_attn<<<8192, 256, 0, stream>>>(qT, kT, S);

  k_softmax<<<BATCH * HWSZ, 256, 0, stream>>>(S);

  k_gemm_o<<<1024, 256, 0, stream>>>(vR, S, oT);

  k_gemm_proj<<<1024, 256, 0, stream>>>(Wpb, oT, bp, x, out);
}
